// HeterogeneousGraphTransformer_71588514890089
// MI455X (gfx1250) — matrix operand order repaired and checked by reading, not compiled, not run on hardware
//
#include <hip/hip_runtime.h>
#include <hip/hip_bf16.h>
#include <math.h>
#include <stdint.h>

// ---------------- problem constants (match reference) ----------------
#define NN   50000
#define EE   150000
#define CIN  128
#define HID  256
#define COUT 128
#define NH   4
#define HD   64
#define NL   2
#define NT   2
#define NR   3

typedef __attribute__((ext_vector_type(16))) _Float16 v16h;
typedef __attribute__((ext_vector_type(8)))  _Float16 v8h;
typedef __attribute__((ext_vector_type(8)))  float    v8f;
typedef __attribute__((ext_vector_type(4)))  uint32_t u32x4;
typedef __attribute__((ext_vector_type(8)))  int      i32x8;
typedef __attribute__((ext_vector_type(4)))  int      i32x4;

// =====================================================================
// WMMA GEMM:  C[M x Nc] = A16[M x K] * W[K x Nc] + bias
//   A16: f16, row stride lda (halfs)
//   Wt:  f16, PRE-TRANSPOSED [Nc x K] contiguous rows
// Block = 4 waves. Block covers a 64-column strip: the 64 x K B-panel is
// DMA'd into LDS once via the Tensor Data Mover (tensor_load_to_lds,
// TENSORcnt). Each wave computes a 16x64 strip: one A fragment reused by
// 4 v_wmma_f32_16x16x32_f16 per K-step, B fragments via ds_load_b128.
// M must be a multiple of 16 (50000 = 3125*16); Nc a multiple of 64.
// =====================================================================
__global__ __launch_bounds__(128)
void hgt_wmma_gemm(const _Float16* __restrict__ A, int lda,
                   const _Float16* __restrict__ Wt,
                   const float* __restrict__ bias,
                   float* __restrict__ Cf, _Float16* __restrict__ Ch, int ldc,
                   int Mtiles, int K)
{
    __shared__ __align__(16) _Float16 ldsB[64 * 256];   // up to K=256
    const int lane = threadIdx.x & 31;
    const int wave = threadIdx.x >> 5;
    const int n0   = blockIdx.x * 64;

    // ---- TDM: DMA 64 rows x K halfs of Wt (rows n0..n0+63) into LDS ----
    if (wave == 0) {
        uint32_t ldsb = (uint32_t)(uintptr_t)(void*)ldsB;     // LDS byte addr
        uint64_t ga   = (uint64_t)(uintptr_t)(Wt + (size_t)n0 * K);
        u32x4 g0 = {0u, 0u, 0u, 0u};
        g0.x = 1u;                                            // count = 1
        g0.y = ldsb;                                          // lds_addr
        g0.z = (uint32_t)ga;                                  // global_addr lo
        g0.w = (uint32_t)((ga >> 32) & 0x01FFFFFFu) | (2u << 30); // hi | type=2
        i32x8 g1 = {0, 0, 0, 0, 0, 0, 0, 0};
        g1[0] = 0x00010000;                  // data_size = 2 bytes
        g1[1] = (int)((unsigned)(K & 0xFFFF) << 16);   // tensor_dim0[15:0]
        g1[2] = (int)(64u << 16);            // tensor_dim0 hi | tensor_dim1 = 64
        g1[3] = (int)((unsigned)(K & 0xFFFF) << 16);   // tdim1 hi | tile_dim0 = K
        g1[4] = 64;                          // tile_dim1 = 64, tile_dim2 = 0
        g1[5] = (int)K;                      // tensor_dim0_stride = K
        g1[6] = 0;
        g1[7] = 0;
        i32x4 z4 = {0, 0, 0, 0};
        i32x8 z8 = {0, 0, 0, 0, 0, 0, 0, 0};
        __builtin_amdgcn_tensor_load_to_lds(g0, g1, z4, z4, z8, 0);
        __builtin_amdgcn_s_wait_tensorcnt(0);                 // s_wait_tensorcnt
    }
    __syncthreads();

    const int tm = blockIdx.y * 4 + wave;
    if (tm >= Mtiles) return;                 // wave-uniform: EXEC stays full
    const int m0 = tm * 16;
    const int r  = lane & 15;
    const int hi = (lane >> 4) & 1;

    // A fragment (16-bit A 16x32 layout): lane<16 row=r holds K {0..7,16..23},
    // lane>=16 holds K {8..15,24..31} -> base offset hi*8, second chunk +16.
    const _Float16* arow = A + (size_t)(m0 + r) * lda + hi * 8;

    v8f acc[4] = {v8f{}, v8f{}, v8f{}, v8f{}};
    for (int kk = 0; kk < K; kk += 32) {
        __builtin_prefetch(arow + kk + 128, 0, 1);   // global_prefetch_b8
        union { v16h v; v8h h[2]; } a;
        a.h[0] = *(const v8h*)(arow + kk);
        a.h[1] = *(const v8h*)(arow + kk + 16);
#pragma unroll
        for (int j = 0; j < 4; ++j) {
            // B fragment (16-bit B 32x16): lanes 0-15 hold K {0..7,16..23} of
            // col n0+j*16+r, lanes 16-31 hold K {8..15,24..31} (ds_load_b128).
            union { v16h v; v8h h[2]; } b;
            const _Float16* brow = ldsB + (size_t)(j * 16 + r) * K + hi * 8 + kk;
            b.h[0] = *(const v8h*)(brow);
            b.h[1] = *(const v8h*)(brow + 16);
            acc[j] = __builtin_amdgcn_wmma_f32_16x16x32_f16(
                         false, a.v, false, b.v, (short)0, acc[j], false, false);
        }
    }

    // C/D layout: lane<16 -> col base+lane, rows m0+0..7 ; lane>=16 -> m0+8..15
    const int rowb = m0 + hi * 8;
#pragma unroll
    for (int j = 0; j < 4; ++j) {
        const int col = n0 + j * 16 + r;
        const float bv = bias ? bias[col] : 0.0f;
#pragma unroll
        for (int i = 0; i < 8; ++i) {
            float val = acc[j][i] + bv;
            if (Cf) Cf[(size_t)(rowb + i) * ldc + col] = val;
            if (Ch) Ch[(size_t)(rowb + i) * ldc + col] = (_Float16)val;
        }
    }
}

// ---------------- helpers ----------------
__global__ void hgt_cvt_f32_f16(const float* __restrict__ in,
                                _Float16* __restrict__ out, int n)
{
    int i = blockIdx.x * blockDim.x + threadIdx.x;
    if (i < n) out[i] = (_Float16)in[i];
}

// W [K x Nc] f32 row-major  ->  Wt [Nc x K] f16
__global__ void hgt_transpose_w(const float* __restrict__ W,
                                _Float16* __restrict__ Wt, int K, int Nc)
{
    int i = blockIdx.x * blockDim.x + threadIdx.x;
    if (i >= K * Nc) return;
    int kk = i / Nc, nn = i % Nc;
    Wt[(size_t)nn * K + kk] = (_Float16)W[i];
}

__global__ void hgt_init_softmax(float* __restrict__ mx,
                                 float* __restrict__ den, int n)
{
    int i = blockIdx.x * blockDim.x + threadIdx.x;
    if (i < n) { mx[i] = -INFINITY; den[i] = 0.0f; }
}

// ---------------- edge phase ----------------
// wave per edge: score[e,h] = (q[dst] . k_rel[src]) * p_rel[h] * scale ;
// atomic float-max into mx[dst,h]
__global__ __launch_bounds__(256)
void hgt_edge_score(const int* __restrict__ src, const int* __restrict__ dst,
                    const _Float16* __restrict__ q16,   // [N,256] (dst type)
                    const float* __restrict__ krel,     // [N,256]
                    const float* __restrict__ prel,     // [H]
                    float* __restrict__ score,          // [E,H]
                    float* __restrict__ mx,             // [N,H]
                    int E, float scale)
{
    int w    = (blockIdx.x * blockDim.x + threadIdx.x) >> 5;
    int lane = threadIdx.x & 31;
    if (w >= E) return;
    int s = src[w], d = dst[w];
    const _Float16* qrow = q16  + (size_t)d * HID;
    const float*    krow = krel + (size_t)s * HID;
#pragma unroll
    for (int h = 0; h < NH; ++h) {
        float p = (float)qrow[h * HD + lane]      * krow[h * HD + lane]
                + (float)qrow[h * HD + 32 + lane] * krow[h * HD + 32 + lane];
#pragma unroll
        for (int off = 16; off; off >>= 1) p += __shfl_xor(p, off, 32);
        if (lane == 0) {
            float sc = p * prel[h] * scale;
            score[(size_t)w * NH + h] = sc;
            float* addr = mx + (size_t)d * NH + h;   // float atomic-max trick
            if (sc >= 0.0f) atomicMax((int*)addr, __float_as_int(sc));
            else            atomicMin((unsigned int*)addr, __float_as_uint(sc));
        }
    }
}

// thread per (e,h): ex = exp(score - mx[dst]); den[dst,h] += ex
__global__ void hgt_edge_exp(const int* __restrict__ dst,
                             float* __restrict__ score,
                             const float* __restrict__ mx,
                             float* __restrict__ den, int EH)
{
    int i = blockIdx.x * blockDim.x + threadIdx.x;
    if (i >= EH) return;
    int e = i >> 2, h = i & 3;
    size_t di = (size_t)dst[e] * NH + h;
    float m = mx[di];
    if (!(m > -3.0e38f)) m = 0.0f;               // isfinite fallback
    float ex = expf(score[i] - m);
    score[i] = ex;
    atomicAdd(den + di, ex);
}

// wave per edge: agg[dst] += v_rel[src] * (ex/den[dst])
__global__ __launch_bounds__(256)
void hgt_edge_aggregate(const int* __restrict__ src, const int* __restrict__ dst,
                        const float* __restrict__ ex,   // [E,H]
                        const float* __restrict__ den,  // [N,H]
                        const float* __restrict__ vrel, // [N,256]
                        float* __restrict__ agg,        // [N,256] (dst type)
                        int E)
{
    int w    = (blockIdx.x * blockDim.x + threadIdx.x) >> 5;
    int lane = threadIdx.x & 31;
    if (w >= E) return;
    int s = src[w], d = dst[w];
    const float* vrow = vrel + (size_t)s * HID;
    float*       arow = agg  + (size_t)d * HID;
#pragma unroll
    for (int h = 0; h < NH; ++h) {
        float attn = ex[(size_t)w * NH + h] / den[(size_t)d * NH + h];
        atomicAdd(arow + h * HD + lane,      vrow[h * HD + lane]      * attn);
        atomicAdd(arow + h * HD + 32 + lane, vrow[h * HD + 32 + lane] * attn);
    }
}

// exact GELU -> f16 (feeds the W_a WMMA GEMM)
__global__ void hgt_gelu_f16(const float* __restrict__ in,
                             _Float16* __restrict__ out, int n)
{
    int i = blockIdx.x * blockDim.x + threadIdx.x;
    if (i >= n) return;
    float x = in[i];
    out[i] = (_Float16)(0.5f * x * (1.0f + erff(x * 0.70710678118654752f)));
}

// wave per row: h = (2-beta)*h + beta*conv ; LayerNorm ; GELU ; write f32+f16
__global__ __launch_bounds__(256)
void hgt_residual_ln(float* __restrict__ h, _Float16* __restrict__ h16,
                     const float* __restrict__ conv,     // [T*N,256]
                     const float* __restrict__ skip_l,   // [T]
                     const float* __restrict__ gamma,    // [T,256]
                     const float* __restrict__ betap,    // [T,256]
                     int rows)
{
    int w    = (blockIdx.x * blockDim.x + threadIdx.x) >> 5;
    int lane = threadIdx.x & 31;
    if (w >= rows) return;
    int t = (w >= NN) ? 1 : 0;
    float sb = 1.0f / (1.0f + expf(-skip_l[t]));
    float*       hr = h    + (size_t)w * HID;
    const float* cr = conv + (size_t)w * HID;

    float vals[8], sum = 0.0f;
#pragma unroll
    for (int i = 0; i < 8; ++i) {
        int c = lane + i * 32;
        float v = (2.0f - sb) * hr[c] + sb * cr[c];
        vals[i] = v; sum += v;
    }
#pragma unroll
    for (int off = 16; off; off >>= 1) sum += __shfl_xor(sum, off, 32);
    float mu = sum * (1.0f / 256.0f);
    float vs = 0.0f;
#pragma unroll
    for (int i = 0; i < 8; ++i) { float dd = vals[i] - mu; vs += dd * dd; }
#pragma unroll
    for (int off = 16; off; off >>= 1) vs += __shfl_xor(vs, off, 32);
    float inv = rsqrtf(vs * (1.0f / 256.0f) + 1e-5f);
#pragma unroll
    for (int i = 0; i < 8; ++i) {
        int c = lane + i * 32;
        float y = (vals[i] - mu) * inv * gamma[t * HID + c] + betap[t * HID + c];
        float g = 0.5f * y * (1.0f + erff(y * 0.70710678118654752f));
        hr[c] = g;
        h16[(size_t)w * HID + c] = (_Float16)g;
    }
}

// =====================================================================
extern "C" void kernel_launch(void* const* d_in, const int* in_sizes, int n_in,
                              void* d_out, int out_size, void* d_ws, size_t ws_size,
                              hipStream_t stream)
{
    (void)in_sizes; (void)n_in; (void)out_size; (void)ws_size;
    const float* x_gene = (const float*)d_in[0];
    const float* x_prot = (const float*)d_in[1];
    const float* W_in   = (const float*)d_in[2];
    const float* b_in   = (const float*)d_in[3];
    const float* W_k    = (const float*)d_in[4];
    const float* b_k    = (const float*)d_in[5];
    const float* W_q    = (const float*)d_in[6];
    const float* b_q    = (const float*)d_in[7];
    const float* W_v    = (const float*)d_in[8];
    const float* b_v    = (const float*)d_in[9];
    const float* W_a    = (const float*)d_in[10];
    const float* b_a    = (const float*)d_in[11];
    const float* skip   = (const float*)d_in[12];
    const float* a_rel  = (const float*)d_in[13];
    const float* m_rel  = (const float*)d_in[14];
    const float* p_rel  = (const float*)d_in[15];
    const float* ln_g   = (const float*)d_in[16];
    const float* ln_b   = (const float*)d_in[17];
    const float* W_out  = (const float*)d_in[18];
    const float* b_out  = (const float*)d_in[19];
    const int* edges[NR] = { (const int*)d_in[20], (const int*)d_in[21],
                             (const int*)d_in[22] };
    float* out = (float*)d_out;

    // ---- workspace bump allocator ----
    char* ws = (char*)d_ws;
    size_t off = 0;
    auto alloc = [&](size_t bytes) -> void* {
        off = (off + 255) & ~(size_t)255;
        void* p = ws + off; off += bytes; return p;
    };
    _Float16* x16  = (_Float16*)alloc((size_t)NT * NN * CIN * 2);
    float*    h    = (float*)   alloc((size_t)NT * NN * HID * 4);
    _Float16* h16  = (_Float16*)alloc((size_t)NT * NN * HID * 2);
    _Float16* k16  = (_Float16*)alloc((size_t)NT * NN * HID * 2);
    _Float16* q16  = (_Float16*)alloc((size_t)NT * NN * HID * 2);
    _Float16* v16  = (_Float16*)alloc((size_t)NT * NN * HID * 2);
    float*    krel = (float*)   alloc((size_t)NN * HID * 4);
    float*    vrel = (float*)   alloc((size_t)NN * HID * 4);
    float*    agg  = (float*)   alloc((size_t)NT * NN * HID * 4);
    _Float16* o16  = (_Float16*)alloc((size_t)NT * NN * HID * 2);
    float*    score= (float*)   alloc((size_t)EE * NH * 4);
    float*    mx   = (float*)   alloc((size_t)NN * NH * 4);
    float*    den  = (float*)   alloc((size_t)NN * NH * 4);
    _Float16* winT = (_Float16*)alloc((size_t)NT * HID * CIN * 2);
    _Float16* wkT  = (_Float16*)alloc((size_t)NL * NT * HID * HID * 2);
    _Float16* wqT  = (_Float16*)alloc((size_t)NL * NT * HID * HID * 2);
    _Float16* wvT  = (_Float16*)alloc((size_t)NL * NT * HID * HID * 2);
    _Float16* waT  = (_Float16*)alloc((size_t)NL * NT * HID * HID * 2);
    _Float16* woT  = (_Float16*)alloc((size_t)NT * COUT * HID * 2);
    _Float16* arT  = (_Float16*)alloc((size_t)NL * NR * NH * HD * HD * 2);
    _Float16* mrT  = (_Float16*)alloc((size_t)NL * NR * NH * HD * HD * 2);

    auto tpose = [&](const float* W, _Float16* Wt, int K, int Nc) {
        int total = K * Nc;
        hgt_transpose_w<<<(total + 255) / 256, 256, 0, stream>>>(W, Wt, K, Nc);
    };
    auto gemm = [&](const _Float16* A, int lda, const _Float16* Wt,
                    const float* bias, float* Cf, _Float16* Ch, int ldc,
                    int M, int Nc, int K) {
        dim3 grid(Nc / 64, (M / 16 + 3) / 4);
        hgt_wmma_gemm<<<grid, 128, 0, stream>>>(A, lda, Wt, bias, Cf, Ch, ldc,
                                                M / 16, K);
    };

    // ---- one-time weight transposes (f32 -> f16^T) ----
    for (int t = 0; t < NT; ++t) {
        tpose(W_in  + (size_t)t * CIN * HID, winT + (size_t)t * HID * CIN, CIN, HID);
        tpose(W_out + (size_t)t * HID * COUT, woT + (size_t)t * COUT * HID, HID, COUT);
    }
    for (int l = 0; l < NL; ++l)
        for (int t = 0; t < NT; ++t) {
            size_t o4 = (size_t)(l * NT + t) * HID * HID;
            tpose(W_k + o4, wkT + o4, HID, HID);
            tpose(W_q + o4, wqT + o4, HID, HID);
            tpose(W_v + o4, wvT + o4, HID, HID);
            tpose(W_a + o4, waT + o4, HID, HID);
        }
    for (int l = 0; l < NL; ++l)
        for (int e = 0; e < NR; ++e)
            for (int hh = 0; hh < NH; ++hh) {
                size_t o4 = (size_t)((l * NR + e) * NH + hh) * HD * HD;
                tpose(a_rel + o4, arT + o4, HD, HD);
                tpose(m_rel + o4, mrT + o4, HD, HD);
            }

    // ---- input projections: h = x @ W_in + b_in (f32 + f16 copies) ----
    hgt_cvt_f32_f16<<<(NN * CIN + 255) / 256, 256, 0, stream>>>(x_gene, x16, NN * CIN);
    hgt_cvt_f32_f16<<<(NN * CIN + 255) / 256, 256, 0, stream>>>(x_prot, x16 + (size_t)NN * CIN, NN * CIN);
    for (int t = 0; t < NT; ++t)
        gemm(x16 + (size_t)t * NN * CIN, CIN, winT + (size_t)t * HID * CIN,
             b_in + (size_t)t * HID,
             h + (size_t)t * NN * HID, h16 + (size_t)t * NN * HID, HID,
             NN, HID, CIN);

    const int EST[NR] = {0, 0, 1};   // src node type per relation
    const int EDT[NR] = {0, 1, 1};   // dst node type per relation
    const float scale = 0.125f;      // 1/sqrt(64)

    for (int l = 0; l < NL; ++l) {
        // K/Q/V projections -> f16
        for (int t = 0; t < NT; ++t) {
            size_t wo = (size_t)(l * NT + t) * HID * HID;
            size_t bo = (size_t)(l * NT + t) * HID;
            size_t no = (size_t)t * NN * HID;
            gemm(h16 + no, HID, wkT + wo, b_k + bo, nullptr, k16 + no, HID, NN, HID, HID);
            gemm(h16 + no, HID, wqT + wo, b_q + bo, nullptr, q16 + no, HID, NN, HID, HID);
            gemm(h16 + no, HID, wvT + wo, b_v + bo, nullptr, v16 + no, HID, NN, HID, HID);
        }
        (void)hipMemsetAsync(agg, 0, (size_t)NT * NN * HID * 4, stream);

        for (int e = 0; e < NR; ++e) {
            const int st = EST[e], dt = EDT[e];
            const int* src = edges[e];
            const int* dst = edges[e] + EE;
            // per-head relation transforms (WMMA GEMMs, K=64, Nc=64)
            for (int hh = 0; hh < NH; ++hh) {
                size_t ro = (size_t)((l * NR + e) * NH + hh) * HD * HD;
                gemm(k16 + (size_t)st * NN * HID + hh * HD, HID, arT + ro,
                     nullptr, krel + hh * HD, nullptr, HID, NN, HD, HD);
                gemm(v16 + (size_t)st * NN * HID + hh * HD, HID, mrT + ro,
                     nullptr, vrel + hh * HD, nullptr, HID, NN, HD, HD);
            }
            hgt_init_softmax<<<(NN * NH + 255) / 256, 256, 0, stream>>>(mx, den, NN * NH);
            hgt_edge_score<<<(EE * 32 + 255) / 256, 256, 0, stream>>>(
                src, dst, q16 + (size_t)dt * NN * HID, krel,
                p_rel + (size_t)(l * NR + e) * NH, score, mx, EE, scale);
            hgt_edge_exp<<<(EE * NH + 255) / 256, 256, 0, stream>>>(
                dst, score, mx, den, EE * NH);
            hgt_edge_aggregate<<<(EE * 32 + 255) / 256, 256, 0, stream>>>(
                src, dst, score, den, vrel, agg + (size_t)dt * NN * HID, EE);
        }

        // epilogue: gelu -> W_a GEMM (writes conv into agg) -> skip/LN/gelu
        hgt_gelu_f16<<<(NT * NN * HID + 255) / 256, 256, 0, stream>>>(
            agg, o16, NT * NN * HID);
        for (int t = 0; t < NT; ++t) {
            size_t wo = (size_t)(l * NT + t) * HID * HID;
            size_t no = (size_t)t * NN * HID;
            gemm(o16 + no, HID, waT + wo, b_a + (size_t)(l * NT + t) * HID,
                 agg + no, nullptr, HID, NN, HID, HID);
        }
        hgt_residual_ln<<<(NT * NN * 32 + 255) / 256, 256, 0, stream>>>(
            h, h16, agg, skip + (size_t)l * NT, ln_g, ln_b, NT * NN);
    }

    // ---- output projection ----
    for (int t = 0; t < NT; ++t)
        gemm(h16 + (size_t)t * NN * HID, HID, woT + (size_t)t * COUT * HID,
             b_out + (size_t)t * COUT,
             out + (size_t)t * NN * COUT, nullptr, COUT, NN, COUT, HID);
}
